// RNNSampleModel_66477503808050
// MI455X (gfx1250) — matrix operand order repaired and checked by reading, not compiled, not run on hardware
//
#include <hip/hip_runtime.h>
#include <hip/hip_bf16.h>
#include <math.h>

#define NUM_SEQ 4
#define BATCH   256
#define TSTEPS  64
#define FEAT    32
#define SDIM    512
#define KDIM    (FEAT + SDIM)   // 544
#define KSLABS  (KDIM / 32)     // 17
#define A_PAD   552             // 544 + 8 bf16 pad -> conflict-free b128 LDS reads

typedef __attribute__((ext_vector_type(16))) __bf16 v16bf;
typedef __attribute__((ext_vector_type(8)))  float  v8f;
typedef __hip_bfloat16 bf16;

// ---------------- prep: inputs f32 [4][B][T][F] -> bf16 [4][T][B][F] ----------------
__global__ void prep_x(const float* __restrict__ in, bf16* __restrict__ out) {
    int o = blockIdx.x * blockDim.x + threadIdx.x;
    if (o >= NUM_SEQ * TSTEPS * BATCH * FEAT) return;
    int f = o & (FEAT - 1);
    int r = o >> 5;
    int b = r & (BATCH - 1); r >>= 8;
    int t = r & (TSTEPS - 1);
    int i = r >> 6;
    out[o] = __float2bfloat16(in[(((i * BATCH + b) * TSTEPS) + t) * FEAT + f]);
}

// ---- prep: W_cell f32 [4][544][512] -> bf16 fragment-major [4][17][ntile32][lane32][e16]
// B-frag k order used here (16-bit, 32x16): lane L holds col N=L%16, K=(L/16)*16+e
__global__ void prep_w(const float* __restrict__ W, bf16* __restrict__ out) {
    int o = blockIdx.x * blockDim.x + threadIdx.x;
    const int TOT = NUM_SEQ * KSLABS * 32 * 32 * 16;
    if (o >= TOT) return;
    int e = o & 15;
    int r = o >> 4;
    int lane  = r & 31; r >>= 5;
    int ntile = r & 31; r >>= 5;
    int kk = r % KSLABS;
    int i  = r / KSLABS;
    int K = kk * 32 + (lane >> 4) * 16 + e;
    int N = ntile * 16 + (lane & 15);
    out[o] = __float2bfloat16(W[(i * KDIM + K) * SDIM + N]);
}

// ---------------- one recurrent step: H_new = tanh([x_t | hc] @ Wc + bc) ----------------
__global__ __launch_bounds__(256)
void rnn_step(const bf16* __restrict__ Xbf,     // [4][T][B][F] bf16
              const bf16* __restrict__ Bshuf,   // [4][17][32][32][16] bf16
              const bf16* __restrict__ hprev,   // [B][S] lane i state t-1, or null (t==0 -> 0)
              const bf16* __restrict__ ph,      // [B][S] lane i-1 state t, or null (lane 0)
              bf16*       __restrict__ hout,    // [B][S] lane i state t
              const float* __restrict__ Wg,     // [1024] gate weights or null
              const float* __restrict__ bg,     // gate bias ptr or null
              const float* __restrict__ bc,     // [S] cell bias
              int seq, int t)
{
    __shared__ __align__(16) bf16 Atile[32 * A_PAD];
    __shared__ float gpart[32 * 8];
    __shared__ float gw[32];

    const int tid  = threadIdx.x;
    const int bx   = blockIdx.x;          // M group: rows bx*32 ..
    const int by   = blockIdx.y;          // N group: cols by*64 ..
    const int row0 = bx * 32;

    // ---- 1) gate: w = sigmoid([h, ph] @ Wg + bg), 8 partials per row ----
    if (ph != nullptr) {
        int r = tid >> 3, p = tid & 7;
        const bf16* prow = ph + (size_t)(row0 + r) * SDIM;
        const bf16* hrow = hprev ? hprev + (size_t)(row0 + r) * SDIM : nullptr;
        float s = 0.f;
        for (int j = p * 64; j < p * 64 + 64; ++j) {
            s += __bfloat162float(prow[j]) * Wg[SDIM + j];
            if (hrow) s += __bfloat162float(hrow[j]) * Wg[j];
        }
        gpart[r * 8 + p] = s;
    }
    __syncthreads();
    if (ph != nullptr && tid < 32) {
        float s = bg[0];
        for (int p = 0; p < 8; ++p) s += gpart[tid * 8 + p];
        gw[tid] = 1.f / (1.f + __expf(-s));
    }
    __syncthreads();

    // ---- 2) stage A = [x_t | hc] for this block's 32 rows into LDS (bf16) ----
    const bf16* xbase = Xbf + ((size_t)(seq * TSTEPS + t) * BATCH + row0) * FEAT;
    for (int idx = tid; idx < 32 * FEAT; idx += 256) {
        int r = idx >> 5, c = idx & 31;
        Atile[r * A_PAD + c] = xbase[r * FEAT + c];
    }
    for (int idx = tid; idx < 32 * SDIM; idx += 256) {
        int r = idx >> 9, c = idx & 511;
        float hv = hprev ? __bfloat162float(hprev[(size_t)(row0 + r) * SDIM + c]) : 0.f;
        float hc;
        if (ph != nullptr) {
            float pv = __bfloat162float(ph[(size_t)(row0 + r) * SDIM + c]);
            float w  = gw[r];
            hc = w * hv + (1.f - w) * pv;
        } else {
            hc = hv;
        }
        Atile[r * A_PAD + FEAT + c] = __float2bfloat16(hc);
    }
    __syncthreads();

    // ---- 3) WMMA: each wave owns one 16x16 tile, 17 chained K=32 steps ----
    const int lane  = tid & 31;
    const int wave  = tid >> 5;
    const int wm    = wave >> 2;            // 0..1
    const int wn    = wave & 3;             // 0..3
    const int ntile = by * 4 + wn;          // global 16-col tile (0..31)

    v8f acc = {};
    const int arow = wm * 16 + (lane & 15);
    const int asel = (lane >> 4) << 4;      // lanes 16-31 shift K by 16
    const uint4* bsrc =
        (const uint4*)(Bshuf + (((size_t)(seq * KSLABS) * 32 + ntile) * 32 + lane) * 16);
    // one kk slab = 32 ntiles * 32 lanes * 16 bf16 = 32768 B = 2048 uint4
    for (int kk = 0; kk < KSLABS; ++kk) {
        union { uint4 u[2]; v16bf v; } a, b;
        const bf16* ap = &Atile[arow * A_PAD + kk * 32 + asel];
        a.u[0] = *(const uint4*)(ap);        // K = base + (lane/16)*16 + 0..7
        a.u[1] = *(const uint4*)(ap + 8);    // K = base + (lane/16)*16 + 8..15
        b.u[0] = bsrc[0];
        b.u[1] = bsrc[1];
        bsrc += 2048;
        acc = __builtin_amdgcn_wmma_f32_16x16x32_bf16(
                  false, a.v, false, b.v, (short)0, acc, false, false);
    }

    // ---- 4) epilogue: bias + tanh, store bf16 state ----
    const int colg  = by * 64 + wn * 16 + (lane & 15);
    const float bias = bc[colg];
    const int rbase = row0 + wm * 16 + ((lane >> 4) << 3);
#pragma unroll
    for (int v = 0; v < 8; ++v) {
        float z = acc[v] + bias;
        hout[(size_t)(rbase + v) * SDIM + colg] = __float2bfloat16(tanhf(z));
    }
}

// ---------------- output: out[i*B+b] = dot(state_i_last[b], W_out) ----------------
__global__ void final_out(const bf16* __restrict__ states,   // [4][2][B][S]
                          const float* __restrict__ Wout,    // [S]
                          float* __restrict__ out)
{
    int r = blockIdx.x * blockDim.x + threadIdx.x;
    if (r >= NUM_SEQ * BATCH) return;
    int i = r >> 8;
    int b = r & 255;
    const bf16* st = states + ((size_t)(i * 2 + ((TSTEPS - 1) & 1)) * BATCH + b) * SDIM;
    float s = 0.f;
    for (int c = 0; c < SDIM; ++c)
        s += __bfloat162float(st[c]) * Wout[c];
    out[r] = s;
}

extern "C" void kernel_launch(void* const* d_in, const int* in_sizes, int n_in,
                              void* d_out, int out_size, void* d_ws, size_t ws_size,
                              hipStream_t stream)
{
    (void)in_sizes; (void)n_in; (void)out_size;
    const float* x  = (const float*)d_in[0];   // [4][B][T][F]
    const float* Wc = (const float*)d_in[1];   // [4][544][512]
    const float* bc = (const float*)d_in[2];   // [4][512]
    const float* Wg = (const float*)d_in[3];   // [3][1024][1]
    const float* bg = (const float*)d_in[4];   // [3][1]
    const float* Wo = (const float*)d_in[5];   // [512][1]
    float* out = (float*)d_out;

    const size_t xbytes = (size_t)NUM_SEQ * TSTEPS * BATCH * FEAT * 2;
    const size_t wbytes = (size_t)NUM_SEQ * KSLABS * 32 * 32 * 16 * 2;
    const size_t sbytes = (size_t)NUM_SEQ * 2 * BATCH * SDIM * 2;
    if (ws_size < xbytes + wbytes + sbytes) return;   // ~8.5 MB needed

    char* ws = (char*)d_ws;
    bf16* Xbf = (bf16*)ws;
    bf16* Bs  = (bf16*)(ws + xbytes);
    bf16* St  = (bf16*)(ws + xbytes + wbytes);

    {
        int n = NUM_SEQ * TSTEPS * BATCH * FEAT;
        prep_x<<<(n + 255) / 256, 256, 0, stream>>>(x, Xbf);
        int m = NUM_SEQ * KSLABS * 32 * 32 * 16;
        prep_w<<<(m + 255) / 256, 256, 0, stream>>>(Wc, Bs);
    }

    auto slot = [&](int i, int s) { return St + (size_t)(i * 2 + s) * BATCH * SDIM; };

    dim3 grid(BATCH / 32, SDIM / 64);   // 8 x 8 blocks, 8 waves each
    for (int t = 0; t < TSTEPS; ++t) {
        for (int i = 0; i < NUM_SEQ; ++i) {
            const bf16* hp = (t > 0) ? slot(i, (t - 1) & 1) : nullptr;
            const bf16* ph = (i > 0) ? slot(i - 1, t & 1) : nullptr;
            bf16* ho = slot(i, t & 1);
            const float* wg  = (i > 0) ? Wg + (size_t)(i - 1) * 2 * SDIM : nullptr;
            const float* bgp = (i > 0) ? bg + (i - 1) : nullptr;
            rnn_step<<<grid, 256, 0, stream>>>(Xbf, Bs, hp, ph, ho, wg, bgp,
                                               bc + (size_t)i * SDIM, i, t);
        }
    }

    final_out<<<(NUM_SEQ * BATCH + 255) / 256, 256, 0, stream>>>(St, Wo, out);
}
